// ConditionalRoutedFeedForward_65463891526111
// MI455X (gfx1250) — matrix operand order repaired and checked by reading, not compiled, not run on hardware
//
#include <hip/hip_runtime.h>
#include <hip/hip_bf16.h>
#include <stdint.h>

typedef __bf16 bf16;
typedef __attribute__((ext_vector_type(16))) bf16 v16bf;
typedef __attribute__((ext_vector_type(8)))  float v8f;
typedef __attribute__((ext_vector_type(8)))  unsigned short u16x8;

#define BATCH   4
#define SEQ     8192
#define DIM     1024
#define NTOK    (BATCH*SEQ)      // 32768
#define LHID    512
#define HHID    4096
#define KSEL    1024
#define GTOK    (BATCH*KSEL)     // 4096
#define EPS_R   0.1f
#define CST_R   0.2079441542f    /* EPS * ln(8) */

// ---------------------------------------------------------------- utilities

__global__ __launch_bounds__(256) void k_cvt_bf16(const float* __restrict__ s,
                                                  bf16* __restrict__ d, int n) {
  int i = blockIdx.x * 256 + threadIdx.x;
  if (i < n) d[i] = (bf16)s[i];
}

// per-token routing logit s = x . rt, and inv_norm = 32 / max(||x||, 1e-12)
__global__ __launch_bounds__(256) void k_prep(const float* __restrict__ x,
                                              const float* __restrict__ rt,
                                              float* __restrict__ s_out,
                                              float* __restrict__ inv_out) {
  __shared__ float rd[256], rs[256];
  int tok = blockIdx.x;
  const float* xr = x + (size_t)tok * DIM;
  float dot = 0.f, ss = 0.f;
  for (int e = 0; e < 4; ++e) {
    int i = threadIdx.x + e * 256;
    float v = xr[i];
    dot += v * rt[i];
    ss  += v * v;
  }
  rd[threadIdx.x] = dot; rs[threadIdx.x] = ss;
  __syncthreads();
  for (int off = 128; off > 0; off >>= 1) {
    if (threadIdx.x < off) {
      rd[threadIdx.x] += rd[threadIdx.x + off];
      rs[threadIdx.x] += rs[threadIdx.x + off];
    }
    __syncthreads();
  }
  if (threadIdx.x == 0) {
    s_out[tok]   = rd[0];
    inv_out[tok] = 32.0f / fmaxf(sqrtf(rs[0]), 1e-12f);
  }
}

// ------------------------------------------------- coordinate-descent router
__global__ __launch_bounds__(256) void k_route(const float* __restrict__ s_all,
                                               float* __restrict__ scores) {
  __shared__ float red[256];
  int b = blockIdx.x, tid = threadIdx.x;
  float sv[32], bv[32];
  for (int e = 0; e < 32; ++e) {
    sv[e] = s_all[b * SEQ + tid + e * 256];
    bv[e] = -fmaxf(sv[e], 0.f);          // b0 = -relu(s)
  }
  float a = 0.f;
  for (int it = 0; it < 50; ++it) {
    float mx = -3.4e38f;
    for (int e = 0; e < 32; ++e) mx = fmaxf(mx, (sv[e] + bv[e]) / EPS_R);
    red[tid] = mx; __syncthreads();
    for (int off = 128; off > 0; off >>= 1) {
      if (tid < off) red[tid] = fmaxf(red[tid], red[tid + off]);
      __syncthreads();
    }
    float tmax = red[0]; __syncthreads();
    float sm = 0.f;
    for (int e = 0; e < 32; ++e) sm += expf((sv[e] + bv[e]) / EPS_R - tmax);
    red[tid] = sm; __syncthreads();
    for (int off = 128; off > 0; off >>= 1) {
      if (tid < off) red[tid] += red[tid + off];
      __syncthreads();
    }
    float lse = tmax + logf(red[0]); __syncthreads();
    a = CST_R - EPS_R * lse;
    for (int e = 0; e < 32; ++e) bv[e] = -fmaxf(sv[e] + a, 0.f);
  }
  for (int e = 0; e < 32; ++e)
    scores[b * SEQ + tid + e * 256] = expf((sv[e] + a + bv[e]) / EPS_R);
}

// ---------------------------------------------- stable top-k (bitonic, u64)
// key = (score_bits << 32) | (8191 - idx): desc score, ties -> lowest index
__global__ __launch_bounds__(512) void k_topk(const float* __restrict__ scores,
                                              int* __restrict__ sel) {
  extern __shared__ unsigned long long keys[];
  int b = blockIdx.x, tid = threadIdx.x;
  for (int i = tid; i < SEQ; i += 512) {
    unsigned int sb = __float_as_uint(scores[b * SEQ + i]);  // scores >= 0
    keys[i] = ((unsigned long long)sb << 32) | (unsigned long long)(SEQ - 1 - i);
  }
  __syncthreads();
  for (int k = 2; k <= SEQ; k <<= 1) {
    for (int j = k >> 1; j > 0; j >>= 1) {
      for (int i = tid; i < SEQ; i += 512) {
        int l = i ^ j;
        if (l > i) {
          unsigned long long a = keys[i], c = keys[l];
          bool descBlock = ((i & k) == 0);
          bool sw = descBlock ? (a < c) : (a > c);
          if (sw) { keys[i] = c; keys[l] = a; }
        }
      }
      __syncthreads();
    }
  }
  for (int i = tid; i < KSEL; i += 512)
    sel[b * KSEL + i] = (SEQ - 1) - (int)(keys[i] & 0xFFFFFFFFull);
}

// ------------------------------------- rmsnorm (+gather) to bf16 staging

__global__ __launch_bounds__(256) void k_norm_light(const float* __restrict__ x,
                                                    const float* __restrict__ inv,
                                                    const float* __restrict__ gamma,
                                                    bf16* __restrict__ xn) {
  size_t gid  = (size_t)blockIdx.x * 256 + threadIdx.x;
  size_t base = gid * 4;
  int row = (int)(base >> 10);
  int col = (int)(base & 1023);
  float4 xv = *(const float4*)(x + base);
  float r = inv[row];
  xn[base + 0] = (bf16)(xv.x * r * gamma[col + 0]);
  xn[base + 1] = (bf16)(xv.y * r * gamma[col + 1]);
  xn[base + 2] = (bf16)(xv.z * r * gamma[col + 2]);
  xn[base + 3] = (bf16)(xv.w * r * gamma[col + 3]);
}

__global__ __launch_bounds__(256) void k_gather_heavy(const float* __restrict__ x,
                                                      const float* __restrict__ inv,
                                                      const float* __restrict__ gamma,
                                                      const int* __restrict__ sel,
                                                      bf16* __restrict__ xg) {
  int g = blockIdx.x;
  int b = g >> 10;
  int idx = sel[g];
  size_t srow = (size_t)b * SEQ + idx;
  const float* src = x + srow * DIM;
  float r = inv[srow];
  int col = threadIdx.x * 4;
  float4 xv = *(const float4*)(src + col);
  bf16* dst = xg + (size_t)g * DIM + col;
  dst[0] = (bf16)(xv.x * r * gamma[col + 0]);
  dst[1] = (bf16)(xv.y * r * gamma[col + 1]);
  dst[2] = (bf16)(xv.z * r * gamma[col + 2]);
  dst[3] = (bf16)(xv.w * r * gamma[col + 3]);
}

// --------------------------------------------------- WMMA bf16 tiled GEMM
// Block 128x128, 8 waves (2x4), wave tile 64x32 (4x2 wmma frags), K step 32.
// A in LDS row-major [128][40]; B in LDS TRANSPOSED [n=128][k=40] so both
// fragment types read K-contiguous 16B runs -> ds_load_b128 pairs.
// Global->LDS staging is software-pipelined through registers.
// MODE 0: out_bf16 = gelu(acc + bias)
// MODE 1: out_f32  = acc + bias
// MODE 2: out_f32[b, sel[row], :] += acc + bias   (sel_scores == 1.0 fwd)

__device__ __forceinline__ float gelu_exact(float v) {
  return 0.5f * v * (1.0f + erff(v * 0.70710678118654752f));
}

#define SA 40   // 80B row: 16B-aligned, 20 banks -> conflict-free
#define SB 40

template <int MODE>
__global__ __launch_bounds__(256) void k_gemm(const bf16* __restrict__ A,
                                              const bf16* __restrict__ B,
                                              const float* __restrict__ bias,
                                              bf16* __restrict__ outB,
                                              float* __restrict__ outF,
                                              const int* __restrict__ sel,
                                              int M, int N, int K) {
  __shared__ __align__(16) bf16 As [128 * SA];   // 10 KB
  __shared__ __align__(16) bf16 BsT[128 * SB];   // 10 KB (col-major tile)
  const int tid  = threadIdx.x;
  const int m0   = blockIdx.y * 128;
  const int n0   = blockIdx.x * 128;
  const int wid  = tid >> 5, lane = tid & 31;
  const int wm   = (wid >> 2) * 64;              // 2 waves along M
  const int wn   = (wid & 3) * 32;               // 4 waves along N
  const int hf   = lane >> 4, lr = lane & 15;

  v8f acc[4][2] = {};

  // A staging: 128 rows x 2 chunks of 16 bf16
  const int arow = tid >> 1, achk = tid & 1;
  // B staging: 16 k-pairs x 16 col-groups of 8
  const int kp = tid >> 4, colg = tid & 15;

  const bf16* agp = A + (size_t)(m0 + arow) * K + achk * 16;
  const bf16* bgp0 = B + (size_t)(2 * kp) * N + n0 + colg * 8;
  const bf16* bgp1 = bgp0 + N;

  // preload tile 0 into registers
  uint4 ra  = *(const uint4*)(agp);
  u16x8 rb0 = *(const u16x8*)(bgp0);
  u16x8 rb1 = *(const u16x8*)(bgp1);

  for (int kk = 0; kk < K; kk += 32) {
    // registers -> LDS
    *(uint4*)(&As[arow * SA + achk * 16]) = ra;
#pragma unroll
    for (int c = 0; c < 8; ++c) {
      unsigned int packed = (unsigned int)rb0[c] | ((unsigned int)rb1[c] << 16);
      *(unsigned int*)(&BsT[(colg * 8 + c) * SB + 2 * kp]) = packed;
    }
    __syncthreads();

    // issue next tile's global loads (overlaps with WMMA below)
    if (kk + 32 < K) {
      ra  = *(const uint4*)(agp + kk + 32);
      rb0 = *(const u16x8*)(bgp0 + (size_t)(kk + 32) * N);
      rb1 = *(const u16x8*)(bgp1 + (size_t)(kk + 32) * N);
    }

    // fragments per ISA 7.12.2 (16-bit A 16x32 / B 32x16 layouts, wave32)
    v16bf af[4], bfr[2];
#pragma unroll
    for (int i = 0; i < 4; ++i) {
      const bf16* base = &As[(wm + i * 16 + lr) * SA];
#pragma unroll
      for (int v = 0; v < 8; ++v) {
        int kb = (v < 4) ? (hf * 8 + 2 * v) : (16 + hf * 8 + 2 * (v - 4));
        af[i][2 * v]     = base[kb];
        af[i][2 * v + 1] = base[kb + 1];
      }
    }
#pragma unroll
    for (int j = 0; j < 2; ++j) {
      const bf16* base = &BsT[(wn + j * 16 + lr) * SB];
#pragma unroll
      for (int v = 0; v < 8; ++v) {
        int kb = (v < 4) ? (hf * 8 + 2 * v) : (16 + hf * 8 + 2 * (v - 4));
        bfr[j][2 * v]     = base[kb];
        bfr[j][2 * v + 1] = base[kb + 1];
      }
    }
#pragma unroll
    for (int i = 0; i < 4; ++i)
#pragma unroll
      for (int j = 0; j < 2; ++j)
        acc[i][j] = __builtin_amdgcn_wmma_f32_16x16x32_bf16(
            false, af[i], false, bfr[j], (short)0, acc[i][j], false, false);
    __syncthreads();
  }

  // epilogue: C/D layout -> VGPR v holds row (8*hf + v), lane low nibble = col
#pragma unroll
  for (int i = 0; i < 4; ++i) {
#pragma unroll
    for (int j = 0; j < 2; ++j) {
#pragma unroll
      for (int v = 0; v < 8; ++v) {
        int grow = m0 + wm + i * 16 + hf * 8 + v;
        int gcol = n0 + wn + j * 16 + lr;
        float val = acc[i][j][v] + bias[gcol];
        if (MODE == 0) {
          outB[(size_t)grow * N + gcol] = (bf16)gelu_exact(val);
        } else if (MODE == 1) {
          outF[(size_t)grow * N + gcol] = val;
        } else {
          int b = grow >> 10;                         // gathered row -> batch
          int idx = sel[grow];                        // original token index
          outF[((size_t)b * SEQ + idx) * DIM + gcol] += val;
        }
      }
    }
  }
}

// ------------------------------------------------------------------ driver

extern "C" void kernel_launch(void* const* d_in, const int* in_sizes, int n_in,
                              void* d_out, int out_size, void* d_ws, size_t ws_size,
                              hipStream_t stream) {
  const float* x   = (const float*)d_in[0];
  const float* rt  = (const float*)d_in[1];
  const float* g_l = (const float*)d_in[2];
  const float* w1l = (const float*)d_in[3];
  const float* b1l = (const float*)d_in[4];
  const float* w2l = (const float*)d_in[5];
  const float* b2l = (const float*)d_in[6];
  const float* g_h = (const float*)d_in[7];
  const float* w1h = (const float*)d_in[8];
  const float* b1h = (const float*)d_in[9];
  const float* w2h = (const float*)d_in[10];
  const float* b2h = (const float*)d_in[11];
  float* out = (float*)d_out;
  (void)in_sizes; (void)n_in; (void)out_size; (void)ws_size;

  char* w = (char*)d_ws;
  size_t off = 0;
  auto alloc = [&](size_t bytes) {
    void* p = w + off;
    off += (bytes + 255) & ~(size_t)255;
    return p;
  };
  float* s_buf   = (float*)alloc((size_t)NTOK * 4);
  float* inv_buf = (float*)alloc((size_t)NTOK * 4);
  float* scores  = (float*)alloc((size_t)NTOK * 4);
  int*   sel     = (int*)  alloc((size_t)GTOK * 4);
  bf16*  w1l_b   = (bf16*) alloc((size_t)DIM * LHID * 2);
  bf16*  w2l_b   = (bf16*) alloc((size_t)LHID * DIM * 2);
  bf16*  w1h_b   = (bf16*) alloc((size_t)DIM * HHID * 2);
  bf16*  w2h_b   = (bf16*) alloc((size_t)HHID * DIM * 2);
  bf16*  xn      = (bf16*) alloc((size_t)NTOK * DIM * 2);
  bf16*  hl      = (bf16*) alloc((size_t)NTOK * LHID * 2);
  bf16*  xg      = (bf16*) alloc((size_t)GTOK * DIM * 2);
  bf16*  hh      = (bf16*) alloc((size_t)GTOK * HHID * 2);

  k_cvt_bf16<<<(DIM * LHID + 255) / 256, 256, 0, stream>>>(w1l, w1l_b, DIM * LHID);
  k_cvt_bf16<<<(LHID * DIM + 255) / 256, 256, 0, stream>>>(w2l, w2l_b, LHID * DIM);
  k_cvt_bf16<<<(DIM * HHID + 255) / 256, 256, 0, stream>>>(w1h, w1h_b, DIM * HHID);
  k_cvt_bf16<<<(HHID * DIM + 255) / 256, 256, 0, stream>>>(w2h, w2h_b, HHID * DIM);

  k_prep<<<NTOK, 256, 0, stream>>>(x, rt, s_buf, inv_buf);
  k_route<<<BATCH, 256, 0, stream>>>(s_buf, scores);
  k_topk<<<BATCH, 512, SEQ * 8, stream>>>(scores, sel);
  k_norm_light<<<(NTOK * DIM / 4) / 256, 256, 0, stream>>>(x, inv_buf, g_l, xn);
  k_gather_heavy<<<GTOK, 256, 0, stream>>>(x, inv_buf, g_h, sel, xg);

  // light: [32768,1024]x[1024,512] -> gelu -> [32768,512]x[512,1024] -> out
  k_gemm<0><<<dim3(LHID / 128, NTOK / 128), 256, 0, stream>>>(
      xn, w1l_b, b1l, hl, nullptr, nullptr, NTOK, LHID, DIM);
  k_gemm<1><<<dim3(DIM / 128, NTOK / 128), 256, 0, stream>>>(
      hl, w2l_b, b2l, nullptr, out, nullptr, NTOK, DIM, LHID);
  // heavy: [4096,1024]x[1024,4096] -> gelu -> [4096,4096]x[4096,1024] -> scatter+=
  k_gemm<0><<<dim3(HHID / 128, GTOK / 128), 256, 0, stream>>>(
      xg, w1h_b, b1h, hh, nullptr, nullptr, GTOK, HHID, DIM);
  k_gemm<2><<<dim3(DIM / 128, GTOK / 128), 256, 0, stream>>>(
      hh, w2h_b, b2h, nullptr, out, sel, GTOK, DIM, HHID);
}
